// PropertyPredictor_51118700757156
// MI455X (gfx1250) — compile-verified, matrix operand order checked by reading, not run on hardware
//
#include <hip/hip_runtime.h>
#include <hip/hip_bf16.h>

// ---------------------------------------------------------------------------
// Problem constants (match reference)
// ---------------------------------------------------------------------------
#define NN 50000
#define EE 800000
#define GG 1000
#define DD 128
#define HH 256
#define LL 4
#define PP 3
#define BN_EPS 1e-5f

typedef __bf16 bf16_t;
typedef bf16_t v16bf __attribute__((ext_vector_type(16)));
typedef bf16_t v8bf  __attribute__((ext_vector_type(8)));
typedef float  v8f   __attribute__((ext_vector_type(8)));

__device__ __forceinline__ bf16_t f2bf(float f) {
    // round-to-nearest-even fp32 -> bf16
    unsigned u = __float_as_uint(f);
    unsigned r = (u + 0x7FFFu + ((u >> 16) & 1u)) >> 16;
    unsigned short s = (unsigned short)r;
    bf16_t b;
    __builtin_memcpy(&b, &s, 2);
    return b;
}

// ---------------------------------------------------------------------------
// Utility kernels
// ---------------------------------------------------------------------------
__global__ void fill_f32(float* __restrict__ p, float v, long long n) {
    long long i = (long long)blockIdx.x * blockDim.x + threadIdx.x;
    if (i < n) p[i] = v;
}

__global__ void conv_bf16(const float* __restrict__ in, bf16_t* __restrict__ out,
                          long long n) {
    long long i = (long long)blockIdx.x * blockDim.x + threadIdx.x;
    if (i < n) out[i] = f2bf(in[i]);
}

__global__ void deg_accum(const int* __restrict__ dst, float* __restrict__ deg, int E) {
    int i = blockIdx.x * blockDim.x + threadIdx.x;
    if (i < E) atomicAdd(&deg[dst[i]], 1.0f);
}

__global__ void make_dinv(float* __restrict__ d, int n) {
    int i = blockIdx.x * blockDim.x + threadIdx.x;
    if (i < n) d[i] = rsqrtf(d[i]);
}

// W [K,Nc] fp32 row-major  ->  Wt [Nc,K] bf16 row-major (batched over z)
__global__ void transpose_bf16_k(const float* __restrict__ W, bf16_t* __restrict__ Wt,
                                 int K, int Nc) {
    long long base = (long long)blockIdx.z * K * Nc;
    int idx = blockIdx.x * blockDim.x + threadIdx.x;
    if (idx < K * Nc) {
        int k = idx / Nc, n = idx % Nc;
        Wt[base + (long long)n * K + k] = f2bf(W[base + idx]);
    }
}

// ---------------------------------------------------------------------------
// WMMA GEMM:  C = act(A[M,K]bf16 * W[K,Nc] + bias),  W pre-transposed bf16.
// Block = 256 threads = 8 waves; block tile 128x64; wave tile 16x64 (4 accs).
// B tile (64 cols x 32 K) double-buffered in LDS (2 x 4KB); one barrier per
// k-chunk. sched_group_barrier pins per-chunk order to:
//   3x global load -> 8x ds_load -> 4x wmma -> 1x ds_store
// so the WMMAs issue back-to-back behind a single dscnt wait.
// Optional bf16 shadow output (feeds the next GEMM's A operand).
//
// Fragment layouts per CDNA5 ISA (wave32):
//   A 16x32 bf16: lane&15 = M row; kg=lane>>4 selects K-halves; per lane
//     e=0..7 -> K = k0 + 8*kg + e, e=8..15 -> K = k0 + 16 + 8*kg + (e-8)
//   B 32x16 bf16: mirrored (lane&15 = N col, same K packing)
//   C/D f32 16x16: col = lane&15, row = vgpr + 8*(lane>>4)
// ---------------------------------------------------------------------------
__global__ void __launch_bounds__(256)
gemm_bf16_wmma(const bf16_t* __restrict__ A, const bf16_t* __restrict__ Wt,
               const float* __restrict__ bias, float* __restrict__ C,
               bf16_t* __restrict__ Cbf, int M, int K, int Nc, int doRelu,
               long long aBS, long long wBS, long long biasBS, long long cBS) {
    __shared__ bf16_t Bt[2][64 * 32];   // 2 x 4 KB, double buffered

    A  += (long long)blockIdx.z * aBS;
    Wt += (long long)blockIdx.z * wBS;
    if (bias) bias += (long long)blockIdx.z * biasBS;
    C  += (long long)blockIdx.z * cBS;
    if (Cbf) Cbf += (long long)blockIdx.z * cBS;

    const int tid  = threadIdx.x;
    const int wave = tid >> 5;
    const int lane = tid & 31;
    const int l15  = lane & 15;
    const int kg   = lane >> 4;

    const int rowTile = blockIdx.x * 128 + wave * 16;
    const int colTile = blockIdx.y * 64;
    const int rowA    = rowTile + l15;
    const bool rowOk  = rowA < M;
    const bf16_t* aRow = A + (size_t)rowA * K + 8 * kg;

    // cooperative B-stage mapping: thread -> (row 0..63, K-octet 0..3)
    const int stRow = tid >> 2;
    const int stK   = (tid & 3) * 8;
    const int stOff = stRow * 32 + stK;
    const bf16_t* wRow = Wt + (size_t)(colTile + stRow) * K + stK;

    const int nk = K >> 5;   // number of 32-wide K chunks
    v8f acc[4] = {};

    // stage chunk 0 into buffer 0
    *(v8bf*)&Bt[0][stOff] = *(const v8bf*)wRow;
    __syncthreads();

    for (int kc = 0; kc < nk; ++kc) {
        const int cur = kc & 1;
        const bool haveNext = (kc + 1) < nk;

        // issue global load of next B chunk early (consumed after the WMMAs)
        v8bf wreg = {};
        if (haveNext) wreg = *(const v8bf*)(wRow + (kc + 1) * 32);

        // A fragment for current chunk
        v16bf a;
        if (rowOk) {
            const bf16_t* ap = aRow + kc * 32;
            __builtin_prefetch(ap + 32, 0, 0);   // global_prefetch_b8
            v8bf a0 = ((const v8bf*)ap)[0];      // K = k0+8kg .. +7
            v8bf a1 = ((const v8bf*)ap)[2];      // K = k0+16+8kg .. +7
#pragma unroll
            for (int i = 0; i < 8; ++i) { a[i] = a0[i]; a[8 + i] = a1[i]; }
        } else {
#pragma unroll
            for (int i = 0; i < 16; ++i) a[i] = f2bf(0.0f);
        }

        // load ALL B fragments first (8 x ds_load_b128, one dscnt wait) ...
        v16bf b[4];
#pragma unroll
        for (int t = 0; t < 4; ++t) {
            const bf16_t* lp = &Bt[cur][(t * 16 + l15) * 32 + 8 * kg];
            v8bf b0 = ((const v8bf*)lp)[0];
            v8bf b1 = ((const v8bf*)lp)[2];
#pragma unroll
            for (int i = 0; i < 8; ++i) { b[t][i] = b0[i]; b[t][8 + i] = b1[i]; }
        }
        // ... then issue the 4 WMMAs back-to-back
#pragma unroll
        for (int t = 0; t < 4; ++t) {
            acc[t] = __builtin_amdgcn_wmma_f32_16x16x32_bf16(
                false, a, false, b[t], (short)0, acc[t], false, false);
        }

        // store next chunk into the other buffer (waits for wreg here,
        // overlapped with the WMMAs above), then one barrier
        if (haveNext) *(v8bf*)&Bt[cur ^ 1][stOff] = wreg;

        // pin per-chunk schedule: VMEM reads, DS reads, matrix ops, DS write
        __builtin_amdgcn_sched_group_barrier(0x020, 3, 0);  // global loads
        __builtin_amdgcn_sched_group_barrier(0x100, 8, 0);  // ds_load x8
        __builtin_amdgcn_sched_group_barrier(0x008, 4, 0);  // wmma x4
        __builtin_amdgcn_sched_group_barrier(0x200, 1, 0);  // ds_store

        __syncthreads();
    }

    const int mrow0 = rowTile + 8 * kg;
#pragma unroll
    for (int t = 0; t < 4; ++t) {
        const int col = colTile + t * 16 + l15;
        const float bv = bias ? bias[col] : 0.0f;
#pragma unroll
        for (int r = 0; r < 8; ++r) {
            int row = mrow0 + r;
            if (row < M) {
                float v = acc[t][r] + bv;
                if (doRelu) v = fmaxf(v, 0.0f);
                C[(size_t)row * Nc + col] = v;
                if (Cbf) Cbf[(size_t)row * Nc + col] = f2bf(v);
            }
        }
    }
}

// ---------------------------------------------------------------------------
// Edge scatter: one wave per edge; 32 lanes * 8 feats = 256 features.
// agg[dst] += m[src] * (dinv[src]*dinv[dst])
// ---------------------------------------------------------------------------
__global__ void scatter_edges(const float* __restrict__ m, const int* __restrict__ src,
                              const int* __restrict__ dst, const float* __restrict__ dinv,
                              float* __restrict__ agg, int E) {
    int wave = (blockIdx.x * blockDim.x + threadIdx.x) >> 5;
    int lane = threadIdx.x & 31;
    if (wave >= E) return;
    int s = src[wave], d = dst[wave];
    float en = dinv[s] * dinv[d];
    const float4* mp = (const float4*)(m + (size_t)s * HH);
    float* ap = agg + (size_t)d * HH;
#pragma unroll
    for (int i = 0; i < 2; ++i) {
        int f4 = lane + i * 32;          // float4 index 0..63
        float4 v = mp[f4];
        atomicAdd(ap + f4 * 4 + 0, v.x * en);
        atomicAdd(ap + f4 * 4 + 1, v.y * en);
        atomicAdd(ap + f4 * 4 + 2, v.z * en);
        atomicAdd(ap + f4 * 4 + 3, v.w * en);
    }
}

// ---------------------------------------------------------------------------
// Post-layer: h = relu(BN(agg + m*snorm + b)) + h; also emits bf16 shadow.
// ---------------------------------------------------------------------------
__global__ void post_layer(float* __restrict__ h, bf16_t* __restrict__ hb,
                           const float* __restrict__ m, const float* __restrict__ agg,
                           const float* __restrict__ dinv, const float* __restrict__ b,
                           const float* __restrict__ gamma, const float* __restrict__ beta,
                           const float* __restrict__ mean, const float* __restrict__ var) {
    long long idx = (long long)blockIdx.x * blockDim.x + threadIdx.x;
    if (idx >= (long long)NN * HH) return;
    int n = (int)(idx / HH);
    int f = (int)(idx % HH);
    float di = dinv[n];
    float val = agg[idx] + m[idx] * (di * di) + b[f];
    float hbn = (val - mean[f]) * rsqrtf(var[f] + BN_EPS) * gamma[f] + beta[f];
    float hn = fmaxf(hbn, 0.0f) + h[idx];
    h[idx] = hn;
    hb[idx] = f2bf(hn);
}

// ---------------------------------------------------------------------------
// Global mean pool
// ---------------------------------------------------------------------------
__global__ void pool_sum(const float* __restrict__ h, const int* __restrict__ batch,
                         float* __restrict__ gsum, float* __restrict__ cnt) {
    long long idx = (long long)blockIdx.x * blockDim.x + threadIdx.x;
    if (idx >= (long long)NN * HH) return;
    int n = (int)(idx / HH);
    int f = (int)(idx % HH);
    int b = batch[n];
    atomicAdd(&gsum[(size_t)b * HH + f], h[idx]);
    if (f == 0) atomicAdd(&cnt[b], 1.0f);
}

__global__ void pool_div(bf16_t* __restrict__ gpb, const float* __restrict__ gsum,
                         const float* __restrict__ cnt) {
    int idx = blockIdx.x * blockDim.x + threadIdx.x;
    if (idx >= GG * HH) return;
    gpb[idx] = f2bf(gsum[idx] / fmaxf(cnt[idx / HH], 1.0f));
}

// ---------------------------------------------------------------------------
// Head level 3: out[g*P + p] = dot(z2[p][g], h3_W[p]) + h3_b[p]   (K=64, Nc=1)
// ---------------------------------------------------------------------------
__global__ void head3(const float* __restrict__ z2, const float* __restrict__ w,
                      const float* __restrict__ b, float* __restrict__ out) {
    int idx = blockIdx.x * blockDim.x + threadIdx.x;
    if (idx >= GG * PP) return;
    int g = idx / PP, p = idx % PP;
    const float* z = z2 + ((size_t)p * GG + g) * (HH / 4);
    const float* wp = w + (size_t)p * (HH / 4);
    float acc = b[p];
#pragma unroll
    for (int k = 0; k < HH / 4; ++k) acc += z[k] * wp[k];
    out[(size_t)g * PP + p] = acc;
}

// ---------------------------------------------------------------------------
// Launch
// ---------------------------------------------------------------------------
extern "C" void kernel_launch(void* const* d_in, const int* in_sizes, int n_in,
                              void* d_out, int out_size, void* d_ws, size_t ws_size,
                              hipStream_t stream) {
    (void)in_sizes; (void)n_in; (void)out_size; (void)ws_size;

    const float* x      = (const float*)d_in[0];
    const int*   edge   = (const int*)d_in[1];
    const int*   srcI   = edge;
    const int*   dstI   = edge + EE;
    const int*   batch  = (const int*)d_in[2];
    const float* emb_W  = (const float*)d_in[3];
    const float* emb_b  = (const float*)d_in[4];
    const float* gcn_W  = (const float*)d_in[5];
    const float* gcn_b  = (const float*)d_in[6];
    const float* bn_g   = (const float*)d_in[7];
    const float* bn_b   = (const float*)d_in[8];
    const float* bn_m   = (const float*)d_in[9];
    const float* bn_v   = (const float*)d_in[10];
    const float* h1W    = (const float*)d_in[11];
    const float* h1b    = (const float*)d_in[12];
    const float* h2W    = (const float*)d_in[13];
    const float* h2b    = (const float*)d_in[14];
    const float* h3W    = (const float*)d_in[15];
    const float* h3b    = (const float*)d_in[16];
    float* out = (float*)d_out;

    // --- workspace carving (256B aligned) ---
    char* p = (char*)d_ws;
    auto alloc = [&](size_t bytes) -> void* {
        void* q = (void*)p;
        p += (bytes + 255) & ~(size_t)255;
        return q;
    };
    float*  dinv  = (float*)alloc((size_t)NN * 4);
    float*  h     = (float*)alloc((size_t)NN * HH * 4);
    bf16_t* hb    = (bf16_t*)alloc((size_t)NN * HH * 2);
    float*  m     = (float*)alloc((size_t)NN * HH * 4);
    float*  agg   = (float*)alloc((size_t)NN * HH * 4);
    bf16_t* embWt = (bf16_t*)alloc((size_t)DD * HH * 2);
    bf16_t* gcnWt = (bf16_t*)alloc((size_t)LL * HH * HH * 2);
    bf16_t* h1Wt  = (bf16_t*)alloc((size_t)PP * HH * (HH / 2) * 2);
    bf16_t* h2Wt  = (bf16_t*)alloc((size_t)PP * (HH / 2) * (HH / 4) * 2);
    // xb (bf16 copy of x, N*D) aliases agg: dead before agg is first zeroed
    bf16_t* xb = (bf16_t*)agg;
    // small pooled/head buffers alias the (free after GCN layers) m region
    float*  gsum = m;                                   // G*H
    float*  cnt  = gsum + (size_t)GG * HH;              // G
    bf16_t* gpb  = (bf16_t*)(cnt + ((GG + 63) & ~63));  // G*H bf16
    float*  z1   = (float*)(gpb + (size_t)GG * HH);     // P*G*(H/2) f32
    bf16_t* z1b  = (bf16_t*)(z1 + (size_t)PP * GG * (HH / 2)); // P*G*(H/2) bf16
    float*  z2   = (float*)(z1b + (size_t)PP * GG * (HH / 2)); // P*G*(H/4) f32

    auto nb = [](long long n) { return (unsigned)((n + 255) / 256); };

    // --- degrees & normalization ---
    fill_f32<<<nb(NN), 256, 0, stream>>>(dinv, 1.0f, NN);
    deg_accum<<<nb(EE), 256, 0, stream>>>(dstI, dinv, EE);
    make_dinv<<<nb(NN), 256, 0, stream>>>(dinv, NN);

    // --- weight transpose + bf16 conversion; input conversion ---
    transpose_bf16_k<<<dim3(nb(DD * HH), 1, 1), 256, 0, stream>>>(emb_W, embWt, DD, HH);
    transpose_bf16_k<<<dim3(nb(HH * HH), 1, LL), 256, 0, stream>>>(gcn_W, gcnWt, HH, HH);
    transpose_bf16_k<<<dim3(nb(HH * (HH / 2)), 1, PP), 256, 0, stream>>>(h1W, h1Wt, HH, HH / 2);
    transpose_bf16_k<<<dim3(nb((HH / 2) * (HH / 4)), 1, PP), 256, 0, stream>>>(h2W, h2Wt, HH / 2, HH / 4);
    conv_bf16<<<nb((long long)NN * DD), 256, 0, stream>>>(x, xb, (long long)NN * DD);

    // --- embedding: h = relu(x @ emb_W + emb_b), plus bf16 shadow hb ---
    dim3 gEmb((NN + 127) / 128, HH / 64, 1);
    gemm_bf16_wmma<<<gEmb, 256, 0, stream>>>(xb, embWt, emb_b, h, hb,
                                             NN, DD, HH, 1, 0, 0, 0, 0);

    // --- GCN layers ---
    dim3 gGcn((NN + 127) / 128, HH / 64, 1);
    for (int i = 0; i < LL; ++i) {
        gemm_bf16_wmma<<<gGcn, 256, 0, stream>>>(hb, gcnWt + (size_t)i * HH * HH,
                                                 nullptr, m, nullptr,
                                                 NN, HH, HH, 0, 0, 0, 0, 0);
        fill_f32<<<nb((long long)NN * HH), 256, 0, stream>>>(agg, 0.0f, (long long)NN * HH);
        scatter_edges<<<(EE * 32 + 255) / 256, 256, 0, stream>>>(m, srcI, dstI, dinv, agg, EE);
        post_layer<<<nb((long long)NN * HH), 256, 0, stream>>>(
            h, hb, m, agg, dinv, gcn_b + i * HH, bn_g + i * HH, bn_b + i * HH,
            bn_m + i * HH, bn_v + i * HH);
    }

    // --- global mean pool (bf16 result feeds head GEMM) ---
    fill_f32<<<nb(GG * HH), 256, 0, stream>>>(gsum, 0.0f, GG * HH);
    fill_f32<<<nb(GG), 256, 0, stream>>>(cnt, 0.0f, GG);
    pool_sum<<<nb((long long)NN * HH), 256, 0, stream>>>(h, batch, gsum, cnt);
    pool_div<<<nb(GG * HH), 256, 0, stream>>>(gpb, gsum, cnt);

    // --- property heads (P batched via blockIdx.z) ---
    dim3 g1((GG + 127) / 128, (HH / 2) / 64, PP);
    gemm_bf16_wmma<<<g1, 256, 0, stream>>>(gpb, h1Wt, h1b, z1, z1b,
                                           GG, HH, HH / 2, 1,
                                           0, (long long)(HH / 2) * HH,
                                           HH / 2, (long long)GG * (HH / 2));
    dim3 g2((GG + 127) / 128, (HH / 4) / 64, PP);
    gemm_bf16_wmma<<<g2, 256, 0, stream>>>(z1b, h2Wt, h2b, z2, nullptr,
                                           GG, HH / 2, HH / 4, 1,
                                           (long long)GG * (HH / 2),
                                           (long long)(HH / 4) * (HH / 2),
                                           HH / 4, (long long)GG * (HH / 4));
    head3<<<nb(GG * PP), 256, 0, stream>>>(z2, h3W, h3b, out);
}
